// Encoder_20272245637276
// MI455X (gfx1250) — matrix operand order repaired and checked by reading, not compiled, not run on hardware
//
#include <hip/hip_runtime.h>
#include <hip/hip_bf16.h>

// ---------------- problem constants (from reference) ----------------
#define NN  20000     // nodes
#define EE  320000    // edges
#define DD  512       // input feature dim
#define HH  512       // hidden dim
#define ZZ  128       // latent dim

typedef unsigned short u16;
typedef __attribute__((ext_vector_type(16))) __bf16 v16bf;
typedef __attribute__((ext_vector_type(8)))  float  v8f;
typedef __attribute__((__vector_size__(16))) int vi4;   // int4 as the builtin expects

#define AS1 __attribute__((address_space(1)))
#define AS3 __attribute__((address_space(3)))

#if defined(__has_builtin)
#if __has_builtin(__builtin_amdgcn_global_load_async_to_lds_b128)
#define HAVE_ASYNC_COPY 1
#else
#define HAVE_ASYNC_COPY 0
#endif
#else
#define HAVE_ASYNC_COPY 0
#endif

union FragBF {
    v16bf v;
    u16   u[16];
    uint4 q[2];
};
static_assert(sizeof(v16bf) == 32, "v16bf must be 32 bytes");

__device__ __forceinline__ u16 f32_to_bf16_bits(float f) {
    unsigned int u = __float_as_uint(f);
    unsigned int r = u + 0x7FFFu + ((u >> 16) & 1u);   // round to nearest even
    return (u16)(r >> 16);
}

__device__ __forceinline__ void wait_async_all() {
#if HAVE_ASYNC_COPY
#if __has_builtin(__builtin_amdgcn_s_wait_asynccnt)
    __builtin_amdgcn_s_wait_asynccnt(0);
#else
    asm volatile("s_wait_asynccnt 0x0" ::: "memory");
#endif
#endif
}

// ---------------- small utility kernels ----------------
__global__ void zero_f32(float* __restrict__ p, long n) {
    long i = (long)blockIdx.x * blockDim.x + threadIdx.x;
    if (i < n) p[i] = 0.0f;
}

__global__ void count_deg(const int* __restrict__ dst, float* __restrict__ deg, int e) {
    int i = blockIdx.x * blockDim.x + threadIdx.x;
    if (i < e) atomicAdd(&deg[dst[i]], 1.0f);
}

__global__ void make_dinv(const float* __restrict__ deg, float* __restrict__ dinv, int n) {
    int i = blockIdx.x * blockDim.x + threadIdx.x;
    if (i < n) dinv[i] = rsqrtf(deg[i] + 1.0f);   // self-loop adds 1
}

__global__ void make_ecoef(const int* __restrict__ src, const int* __restrict__ dst,
                           const float* __restrict__ dinv, float* __restrict__ ec, int e) {
    int i = blockIdx.x * blockDim.x + threadIdx.x;
    if (i < e) ec[i] = dinv[src[i]] * dinv[dst[i]];
}

__global__ void cvt_f32_bf16(const float* __restrict__ in, u16* __restrict__ out, long n) {
    long i = (long)blockIdx.x * blockDim.x + threadIdx.x;
    if (i < n) out[i] = f32_to_bf16_bits(in[i]);
}

// Pre-transpose weight to bf16: Wt[c*K + k] = bf16(W[k*Hc + c]).
// Reads coalesced over c; W is small and converted once per GEMM.
__global__ void cvt_transpose_bf16(const float* __restrict__ W, u16* __restrict__ Wt,
                                   int K, int Hc) {
    long i = (long)blockIdx.x * blockDim.x + threadIdx.x;
    if (i >= (long)K * Hc) return;
    int c = (int)(i % Hc);
    int k = (int)(i / Hc);
    Wt[(long)c * K + k] = f32_to_bf16_bits(W[i]);
}

// ---------------- WMMA GEMM: C[M x Hc] = A[M x K] @ W[K x Hc] ----------------
// A row-major bf16 [M x K]; Wt TRANSPOSED bf16 [Hc x K] (Wt[c][k] = W[k][c]); C fp32.
// K % 32 == 0, Hc % 128 == 0. Rows guarded vs M.
// Block: 256 threads = 8 waves; block tile 64(M) x 128(N); wave (wm 0..3, wn 0..1)
// computes 16 x 64 via 4 accumulators -> 4 v_wmma per k-step, A fragment reused 4x.
// Tiles double-buffered in LDS; staged with async global->LDS copies when available.
__global__ __launch_bounds__(256)
void wmma_gemm_bf16(const u16* __restrict__ A, const u16* __restrict__ Wt,
                    float* __restrict__ C, int M, int K, int Hc)
{
    __shared__ u16 As[2][64 * 32];     // A tile  (row stride 32 in k)
    __shared__ u16 Bs[2][128 * 32];    // W tile, column-major rows: Bs[c][k]

    const int tid  = threadIdx.x;
    const int lane = tid & 31;
    const int wave = tid >> 5;
    const int wm   = wave >> 1;            // 0..3  (row group of 16)
    const int wn   = wave & 1;             // 0..1  (col group of 64)
    const int rowBase = blockIdx.y * 64;
    const int colBase = blockIdx.x * 128;

    const int halfsel = (lane < 16) ? 1 : 0;   // 1 = low half lanes
    const int l15     = lane & 15;

    // ---- staging geometry: 256 threads move 2048 (A) + 4096 (B) bf16 per tile ----
    const int aIdx = tid * 8;                       // A: one 16B chunk per thread
    const int aRow = aIdx >> 5;
    const int aCol = aIdx & 31;
    const int gr   = rowBase + aRow;
    const long aOff = (long)gr * K + aCol;
    const int b0Idx = tid * 8;                      // B: two 16B chunks per thread
    const int b1Idx = (tid + 256) * 8;
    const long bOff0 = (long)(colBase + (b0Idx >> 5)) * K + (b0Idx & 31);
    const long bOff1 = (long)(colBase + (b1Idx >> 5)) * K + (b1Idx & 31);

    v8f acc[4];
    #pragma unroll
    for (int f = 0; f < 4; ++f) acc[f] = (v8f){0.f,0.f,0.f,0.f,0.f,0.f,0.f,0.f};

    auto stage = [&](int buf, int k0) {
#if HAVE_ASYNC_COPY
        if (gr < M)
            __builtin_amdgcn_global_load_async_to_lds_b128(
                (AS1 vi4*)(A + aOff + k0), (AS3 vi4*)&As[buf][aIdx], 0, 0);
        __builtin_amdgcn_global_load_async_to_lds_b128(
            (AS1 vi4*)(Wt + bOff0 + k0), (AS3 vi4*)&Bs[buf][b0Idx], 0, 0);
        __builtin_amdgcn_global_load_async_to_lds_b128(
            (AS1 vi4*)(Wt + bOff1 + k0), (AS3 vi4*)&Bs[buf][b1Idx], 0, 0);
#else
        uint4 av = make_uint4(0u, 0u, 0u, 0u);
        if (gr < M) av = *(const uint4*)(A + aOff + k0);
        *(uint4*)&As[buf][aIdx]  = av;
        *(uint4*)&Bs[buf][b0Idx] = *(const uint4*)(Wt + bOff0 + k0);
        *(uint4*)&Bs[buf][b1Idx] = *(const uint4*)(Wt + bOff1 + k0);
#endif
    };

    const int nk = K >> 5;
    stage(0, 0);

    for (int i = 0; i < nk; ++i) {
        const int buf = i & 1;
        wait_async_all();        // this wave's tile writes to LDS are complete
        __syncthreads();         // everyone's are; prior reads of buf^1 are done
        if (i + 1 < nk) stage(buf ^ 1, (i + 1) << 5);

        // A 16x32 fragment: lanes 0-15 -> K {0..7,16..23}; lanes 16-31 -> K {8..15,24..31}
        FragBF a;
        {
            const u16* ap = &As[buf][(wm * 16 + l15) * 32];
            int off = halfsel ? 0 : 8;
            a.q[0] = *(const uint4*)(ap + off);
            a.q[1] = *(const uint4*)(ap + off + 16);
        }
        // B 32x16 fragments: lane n<16 -> column n, K {0..7,16..23}; lane n>=16 -> column n-16, K {8..15,24..31}
        #pragma unroll
        for (int f = 0; f < 4; ++f) {
            FragBF b;
            const u16* bp = &Bs[buf][(wn * 64 + f * 16 + l15) * 32];
            int off = halfsel ? 0 : 8;
            b.q[0] = *(const uint4*)(bp + off);
            b.q[1] = *(const uint4*)(bp + off + 16);
            acc[f] = __builtin_amdgcn_wmma_f32_16x16x32_bf16(false, a.v, false, b.v,
                                                             (short)0, acc[f], false, false);
        }
    }

    // ---- epilogue: C/D layout: VGPR r -> row r (lanes 0-15) / row 8+r (lanes 16-31) ----
    const int rbase = rowBase + wm * 16 + (halfsel ? 0 : 8);
    #pragma unroll
    for (int f = 0; f < 4; ++f) {
        const int col = colBase + wn * 64 + f * 16 + l15;
        #pragma unroll
        for (int r = 0; r < 8; ++r) {
            int row = rbase + r;
            if (row < M) C[(long)row * Hc + col] = acc[f][r];
        }
    }
}

// ---------------- edge propagation: agg[dst] += ec[e] * t[src] ----------------
__global__ void scatter_edges(const float* __restrict__ t, const int* __restrict__ src,
                              const int* __restrict__ dst, const float* __restrict__ ec,
                              float* __restrict__ agg, int e, int F)
{
    long idx = (long)blockIdx.x * blockDim.x + threadIdx.x;
    int  per = F >> 2;
    long total = (long)e * per;
    if (idx >= total) return;
    int ed = (int)(idx / per);
    int f  = (int)(idx % per) << 2;
    int s  = src[ed], d = dst[ed];
    float c = ec[ed];
    const float4 v = *(const float4*)(t + (long)s * F + f);
    float* out = agg + (long)d * F + f;
    atomicAdd(out + 0, v.x * c);
    atomicAdd(out + 1, v.y * c);
    atomicAdd(out + 2, v.z * c);
    atomicAdd(out + 3, v.w * c);
}

// ---------------- combine: out = [relu][l2norm](agg + dinv^2 * t + b) ----------------
// blockDim.x == 128; one block per row; per-thread elems = F/128 (<= 4).
template<bool RELU, bool NORM, bool BF16OUT>
__global__ __launch_bounds__(128)
void combine_kernel(const float* __restrict__ agg, const float* __restrict__ t,
                    const float* __restrict__ bias, const float* __restrict__ dinv,
                    float* __restrict__ outf, u16* __restrict__ outb, int F)
{
    const int row = blockIdx.x;
    float sc = dinv[row];
    sc *= sc;                                   // self-loop coefficient
    const int per = F >> 7;                     // F/128: 4 (H) or 1 (Z)
    float vals[4];
    float ss = 0.0f;
    for (int j = 0; j < per; ++j) {
        int f = threadIdx.x + (j << 7);
        float v = agg[(long)row * F + f] + sc * t[(long)row * F + f] + bias[f];
        if (RELU) v = fmaxf(v, 0.0f);
        vals[j] = v;
        ss += v * v;
    }
    float scale = 1.0f;
    if (NORM) {
        #pragma unroll
        for (int o = 16; o > 0; o >>= 1) ss += __shfl_down(ss, o, 32);
        __shared__ float red[4];                // 128 threads = 4 wave32s
        if ((threadIdx.x & 31) == 0) red[threadIdx.x >> 5] = ss;
        __syncthreads();
        float tot = red[0] + red[1] + red[2] + red[3];
        scale = 1.0f / fmaxf(sqrtf(tot), 1e-12f);   // * SCALING_FACTOR(=1.0)
    }
    for (int j = 0; j < per; ++j) {
        int f = threadIdx.x + (j << 7);
        float v = vals[j] * scale;
        if (BF16OUT) outb[(long)row * F + f] = f32_to_bf16_bits(v);
        else         outf[(long)row * F + f] = v;
    }
}

// ---------------- z = mu + eps * exp(0.5 * log_var) ----------------
__global__ void sample_z(const float* __restrict__ mu, const float* __restrict__ lv,
                         const float* __restrict__ eps, float* __restrict__ z, long n)
{
    long i = (long)blockIdx.x * blockDim.x + threadIdx.x;
    if (i < n) z[i] = mu[i] + eps[i] * expf(0.5f * lv[i]);
}

// ---------------- host-side orchestration ----------------
extern "C" void kernel_launch(void* const* d_in, const int* in_sizes, int n_in,
                              void* d_out, int out_size, void* d_ws, size_t ws_size,
                              hipStream_t stream)
{
    const float* x   = (const float*)d_in[0];
    const int*   ei  = (const int*)  d_in[1];
    const float* eps = (const float*)d_in[2];
    const float* W1  = (const float*)d_in[3];
    const float* b1  = (const float*)d_in[4];
    const float* W2  = (const float*)d_in[5];
    const float* b2  = (const float*)d_in[6];
    const float* Wmu = (const float*)d_in[7];
    const float* bmu = (const float*)d_in[8];
    const float* Wlv = (const float*)d_in[9];
    const float* blv = (const float*)d_in[10];
    (void)in_sizes; (void)n_in; (void)out_size; (void)ws_size;

    const int* src = ei;            // edge_index[0]
    const int* dst = ei + EE;       // edge_index[1]

    // d_out = [ z | mu | log_var ], each NN*ZZ fp32
    float* z_out  = (float*)d_out;
    float* mu_out = z_out + (long)NN * ZZ;
    float* lv_out = mu_out + (long)NN * ZZ;

    // ---- workspace carve-up (256B aligned segments) ----
    char* ws = (char*)d_ws;
    size_t off = 0;
    auto carve = [&](size_t bytes) {
        void* p = ws + off;
        off = (off + bytes + 255) & ~(size_t)255;
        return p;
    };
    float* dinv = (float*)carve((size_t)NN * 4);
    float* ec   = (float*)carve((size_t)EE * 4);
    float* t    = (float*)carve((size_t)NN * HH * 4);   // GEMM output
    float* agg  = (float*)carve((size_t)NN * HH * 4);   // propagation accumulator
    u16*   Abf  = (u16*)  carve((size_t)NN * HH * 2);   // bf16 activations
    u16*   Wbf  = (u16*)  carve((size_t)HH * HH * 2);   // bf16 transposed weights (reused)

    auto zero = [&](float* p, long n) {
        zero_f32<<<(unsigned)((n + 255) / 256), 256, 0, stream>>>(p, n);
    };
    auto gemm = [&](const u16* A, const u16* Wtb, float* C, int M, int K, int Hc) {
        dim3 g(Hc / 128, (M + 63) / 64);
        wmma_gemm_bf16<<<g, 256, 0, stream>>>(A, Wtb, C, M, K, Hc);
    };
    auto cvt = [&](const float* in, u16* out, long n) {
        cvt_f32_bf16<<<(unsigned)((n + 255) / 256), 256, 0, stream>>>(in, out, n);
    };
    auto cvtT = [&](const float* Wsrc, u16* Wtb, int K, int Hc) {
        long n = (long)K * Hc;
        cvt_transpose_bf16<<<(unsigned)((n + 255) / 256), 256, 0, stream>>>(Wsrc, Wtb, K, Hc);
    };
    auto scatter = [&](const float* tin, float* ag, int F) {
        long total = (long)EE * (F >> 2);
        scatter_edges<<<(unsigned)((total + 255) / 256), 256, 0, stream>>>(
            tin, src, dst, ec, ag, EE, F);
    };

    // ---- degrees / normalization coefficients (use agg as deg scratch) ----
    zero(agg, NN);
    count_deg<<<(EE + 255) / 256, 256, 0, stream>>>(dst, agg, EE);
    make_dinv<<<(NN + 255) / 256, 256, 0, stream>>>(agg, dinv, NN);
    make_ecoef<<<(EE + 255) / 256, 256, 0, stream>>>(src, dst, dinv, ec, EE);

    // ---- layer 1: h1 = l2norm(relu(gcn(x, W1, b1))) -> bf16 ----
    cvt(x, Abf, (long)NN * DD);
    cvtT(W1, Wbf, DD, HH);
    gemm(Abf, Wbf, t, NN, DD, HH);
    zero(agg, (long)NN * HH);
    scatter(t, agg, HH);
    combine_kernel<true, true, true><<<NN, 128, 0, stream>>>(
        agg, t, b1, dinv, nullptr, Abf, HH);

    // ---- layer 2: h2 = relu(gcn(h1, W2, b2)) -> bf16 ----
    cvtT(W2, Wbf, HH, HH);
    gemm(Abf, Wbf, t, NN, HH, HH);
    zero(agg, (long)NN * HH);
    scatter(t, agg, HH);
    combine_kernel<true, false, true><<<NN, 128, 0, stream>>>(
        agg, t, b2, dinv, nullptr, Abf, HH);

    // ---- mu head ----
    cvtT(Wmu, Wbf, HH, ZZ);
    gemm(Abf, Wbf, t, NN, HH, ZZ);
    zero(agg, (long)NN * ZZ);
    scatter(t, agg, ZZ);
    combine_kernel<false, false, false><<<NN, 128, 0, stream>>>(
        agg, t, bmu, dinv, mu_out, nullptr, ZZ);

    // ---- log_var head ----
    cvtT(Wlv, Wbf, HH, ZZ);
    gemm(Abf, Wbf, t, NN, HH, ZZ);
    zero(agg, (long)NN * ZZ);
    scatter(t, agg, ZZ);
    combine_kernel<false, false, false><<<NN, 128, 0, stream>>>(
        agg, t, blv, dinv, lv_out, nullptr, ZZ);

    // ---- reparameterized sample ----
    long nz = (long)NN * ZZ;
    sample_z<<<(unsigned)((nz + 255) / 256), 256, 0, stream>>>(mu_out, lv_out, eps, z_out, nz);
}
